// CustomAttention_26242250179076
// MI455X (gfx1250) — matrix operand order repaired and checked by reading, not compiled, not run on hardware
//
#include <hip/hip_runtime.h>
#include <hip/hip_bf16.h>

typedef _Float16 half_t;
typedef __attribute__((ext_vector_type(16))) _Float16 v16h;
typedef __attribute__((ext_vector_type(8)))  _Float16 v8h;
typedef __attribute__((ext_vector_type(4)))  _Float16 v4h;
typedef __attribute__((ext_vector_type(8)))  float    v8f;
typedef __attribute__((ext_vector_type(4)))  float    v4f;

#define B_  4
#define N_  2048
#define C_  1024
#define H_  16
#define D_  64

__device__ __forceinline__ v8f wmma16(v16h a, v16h b, v8f c) {
  return __builtin_amdgcn_wmma_f32_16x16x32_f16(false, a, false, b, (short)0, c, false, false);
}

// A fragment (16x32 f16, ISA 7.12.2): lane holds row r=lane&15; g=lane>>4 selects
// K groups [g*8, g*8+8) and [16+g*8, 16+g*8+8).
__device__ __forceinline__ v16h load_fragA(const half_t* __restrict__ row, int k, int g) {
  const v8h lo = *(const v8h*)(row + k + g * 8);
  const v8h hi = *(const v8h*)(row + k + 16 + g * 8);
  v16h r;
#pragma unroll
  for (int i = 0; i < 8; ++i) { r[i] = lo[i]; r[i + 8] = hi[i]; }
  return r;
}

// B fragment (32x16 f16): lane holds column n=lane&15 (a row of the row-major
// K-major operand); K groups [g*8, g*8+8) and [16+g*8, 16+g*8+8), matching A.
__device__ __forceinline__ v16h load_fragB(const half_t* __restrict__ col, int k, int g) {
  const v8h lo = *(const v8h*)(col + k + g * 8);
  const v8h hi = *(const v8h*)(col + k + 16 + g * 8);
  v16h r;
#pragma unroll
  for (int i = 0; i < 8; ++i) { r[i] = lo[i]; r[i + 8] = hi[i]; }
  return r;
}

// Async global->LDS copy (gfx1250): VDST is a VGPR holding the LDS byte offset
// (low 32 bits of the generic shared-memory address); tracked by ASYNCcnt.
__device__ __forceinline__ void async_copy_b128(half_t* lds_dst, const half_t* gsrc) {
  const unsigned off = (unsigned)(unsigned long long)lds_dst;
  asm volatile("global_load_async_to_lds_b128 %0, %1, off"
               :: "v"(off), "v"(gsrc) : "memory");
}
__device__ __forceinline__ void wait_async0() {
  asm volatile("s_wait_asynccnt 0x0" ::: "memory");
}

// ---------------------------------------------------------------- fp32 -> f16
__global__ __launch_bounds__(256) void cvt_f32_to_f16(const float* __restrict__ src,
                                                      half_t* __restrict__ dst, int n4) {
  const int i = blockIdx.x * blockDim.x + threadIdx.x;
  if (i >= n4) return;
  const v4f v = ((const v4f*)src)[i];
  v4h o;
#pragma unroll
  for (int j = 0; j < 4; ++j) o[j] = (half_t)v[j];
  ((v4h*)dst)[i] = o;
}

// ------------------------------------------------- QKV GEMM + head scatter
// y = x @ Wqkv^T + b ; q scaled by 1/sqrt(D), stored [B,H,N,D];
// k stored [B,H,N,D]; v stored transposed [B,H,D,N] for contiguous PV B-frags.
__global__ __launch_bounds__(128) void qkv_gemm(const half_t* __restrict__ xh,
                                                const half_t* __restrict__ wh,
                                                const float* __restrict__ bias,
                                                half_t* __restrict__ qh,
                                                half_t* __restrict__ kh,
                                                half_t* __restrict__ vth) {
  const int lane = threadIdx.x & 31;
  const int wave = threadIdx.x >> 5;
  const int g = lane >> 4;
  const int r = lane & 15;
  const int NT = (3 * C_) / 32;                 // 96 column tiles
  const int tile = blockIdx.x * 4 + wave;       // 32x32 output tile per wave
  const int tm = (tile / NT) * 32;
  const int tn = (tile % NT) * 32;

  v8f acc[2][2] = {};
  const half_t* a0p = xh + (size_t)(tm + r) * C_;
  const half_t* a1p = xh + (size_t)(tm + 16 + r) * C_;
  const half_t* b0p = wh + (size_t)(tn + r) * C_;
  const half_t* b1p = wh + (size_t)(tn + 16 + r) * C_;
  for (int k = 0; k < C_; k += 32) {
    const v16h a0 = load_fragA(a0p, k, g);
    const v16h a1 = load_fragA(a1p, k, g);
    const v16h b0 = load_fragB(b0p, k, g);
    const v16h b1 = load_fragB(b1p, k, g);
    acc[0][0] = wmma16(a0, b0, acc[0][0]);
    acc[0][1] = wmma16(a0, b1, acc[0][1]);
    acc[1][0] = wmma16(a1, b0, acc[1][0]);
    acc[1][1] = wmma16(a1, b1, acc[1][1]);
  }

#pragma unroll
  for (int mi = 0; mi < 2; ++mi)
#pragma unroll
    for (int ni = 0; ni < 2; ++ni) {
      const v8f a = acc[mi][ni];
      const int col = tn + ni * 16 + r;         // output feature in [0, 3072)
      const float bv = bias[col];
      const int which = col >> 10;              // 0=q 1=k 2=v (tile-uniform)
      const int c = col & (C_ - 1);
      const int hh = c >> 6;
      const int d = c & (D_ - 1);
      const int m0 = tm + mi * 16 + g * 8;      // 8 consecutive rows per lane
      const int bb = m0 >> 11;
      const int nr0 = m0 & (N_ - 1);
      if (which == 2) {                         // V: transposed, contiguous v8h store
        v8h pk;
#pragma unroll
        for (int j = 0; j < 8; ++j) pk[j] = (half_t)(a[j] + bv);
        *(v8h*)(vth + (((size_t)(bb * H_ + hh) * D_ + d) * N_ + nr0)) = pk;
      } else {
        const float sc = (which == 0) ? 0.125f : 1.0f;   // 1/sqrt(64) folded into Q
        half_t* dst = (which == 0 ? qh : kh) + (((size_t)(bb * H_ + hh) * N_ + nr0) * D_ + d);
#pragma unroll
        for (int j = 0; j < 8; ++j) dst[j * D_] = (half_t)((a[j] + bv) * sc);
      }
    }
}

// ------------------------------------------------- flash attention per (b,h)
// All 8 waves of a block share one (b,h); K/V 32-step tiles are staged once per
// block into double-buffered LDS via async copies (ASYNCcnt) instead of being
// fetched 8x through the vector-load path. S/P stay in registers + a 1KB
// per-wave LDS bounce for the C-layout -> A-layout transpose of P.
__global__ __launch_bounds__(256) void flash_attn(const half_t* __restrict__ qh,
                                                  const half_t* __restrict__ kh,
                                                  const half_t* __restrict__ vth,
                                                  const float* __restrict__ bias,
                                                  half_t* __restrict__ aoh) {
  __shared__ half_t kbuf[2][32 * 64];   // [key row][d]
  __shared__ half_t vbuf[2][64 * 32];   // [d][key row]
  __shared__ half_t pbuf[8 * 16 * 32];
  const int tid = threadIdx.x;
  const int lane = tid & 31;
  const int wave = tid >> 5;
  const int g = lane >> 4;
  const int r = lane & 15;
  const int bh = blockIdx.x >> 4;               // 64 (b,h) pairs
  const int qblk = blockIdx.x & 15;             // 16 blocks of 128 q-rows each
  const int b = bh >> 4;
  const int h = bh & 15;
  const int q0 = qblk * 128 + wave * 16;

  const half_t* Q = qh + ((size_t)bh * N_ + q0) * D_;
  const half_t* K = kh + (size_t)bh * N_ * D_;
  const half_t* V = vth + (size_t)bh * D_ * N_;
  const float* Bp = bias + ((size_t)h * N_ + q0) * N_;
  half_t* pb = pbuf + wave * (16 * 32);

  // cooperative stage of one 32-key step: 4KB K tile + 4KB V tile, one b128 each
  const int krow = tid >> 3, kcol = (tid & 7) * 8;   // K: 32 x 64
  const int vrow = tid >> 2, vcol = (tid & 3) * 8;   // V: 64 x 32

  const v16h qf0 = load_fragA(Q + (size_t)r * D_, 0, g);
  const v16h qf1 = load_fragA(Q + (size_t)r * D_, 32, g);

  v8f o0 = {}, o1 = {}, o2 = {}, o3 = {};
  float mrow[8], lrow[8];
#pragma unroll
  for (int j = 0; j < 8; ++j) { mrow[j] = -1e30f; lrow[j] = 0.f; }

  // prologue stage for kt = 0
  async_copy_b128(&kbuf[0][krow * 64 + kcol], K + (size_t)krow * D_ + kcol);
  async_copy_b128(&vbuf[0][vrow * 32 + vcol], V + (size_t)vrow * N_ + vcol);

  for (int kt = 0; kt < N_; kt += 32) {
    const int cur = (kt >> 5) & 1;
    wait_async0();        // own async loads landed
    __syncthreads();      // everyone's async loads landed
    if (kt + 32 < N_) {   // stage next step into the other buffer
      async_copy_b128(&kbuf[cur ^ 1][krow * 64 + kcol],
                      K + (size_t)(kt + 32 + krow) * D_ + kcol);
      async_copy_b128(&vbuf[cur ^ 1][vrow * 32 + vcol],
                      V + (size_t)vrow * N_ + (kt + 32) + vcol);
      // keep the dominant HBM stream (attn_bias, 256MB) running ahead
      __builtin_prefetch(Bp + (size_t)(g * 8) * N_ + kt + 32 + r, 0, 0);
    }

    v8f s0 = {}, s1 = {};
    {
      const half_t* k0 = &kbuf[cur][r * 64];
      const half_t* k1 = &kbuf[cur][(16 + r) * 64];
      s0 = wmma16(qf0, load_fragB(k0, 0, g), s0);
      s0 = wmma16(qf1, load_fragB(k0, 32, g), s0);
      s1 = wmma16(qf0, load_fragB(k1, 0, g), s1);
      s1 = wmma16(qf1, load_fragB(k1, 32, g), s1);
    }
    // C-layout: element j of lane holds S[m = j + g*8][n = r(+16)]; row stats
    // therefore reduce across the 16-lane group (xor masks 1,2,4,8 stay inside).
    float alpha[8];
#pragma unroll
    for (int j = 0; j < 8; ++j) {
      const float* brow = Bp + (size_t)(j + g * 8) * N_ + kt + r;
      float e0 = s0[j] + brow[0];
      float e1 = s1[j] + brow[16];
      float mx = fmaxf(e0, e1);
      mx = fmaxf(mx, __shfl_xor(mx, 1, 32));
      mx = fmaxf(mx, __shfl_xor(mx, 2, 32));
      mx = fmaxf(mx, __shfl_xor(mx, 4, 32));
      mx = fmaxf(mx, __shfl_xor(mx, 8, 32));
      const float mnew = fmaxf(mrow[j], mx);
      alpha[j] = __expf(mrow[j] - mnew);
      mrow[j] = mnew;
      e0 = __expf(e0 - mnew);
      e1 = __expf(e1 - mnew);
      float rs = e0 + e1;
      rs += __shfl_xor(rs, 1, 32);
      rs += __shfl_xor(rs, 2, 32);
      rs += __shfl_xor(rs, 4, 32);
      rs += __shfl_xor(rs, 8, 32);
      lrow[j] = lrow[j] * alpha[j] + rs;
      const int m = j + g * 8;                  // LDS bounce: write P row-major
      pb[m * 32 + r] = (half_t)e0;              // (per-wave region; DS ops are
      pb[m * 32 + 16 + r] = (half_t)e1;         //  in-order within a wave)
    }
#pragma unroll
    for (int j = 0; j < 8; ++j) {
      o0[j] *= alpha[j]; o1[j] *= alpha[j]; o2[j] *= alpha[j]; o3[j] *= alpha[j];
    }
    const v16h pf = load_fragA(pb + r * 32, 0, g);   // P as A-fragment (16x32)
    o0 = wmma16(pf, load_fragB(&vbuf[cur][(0 * 16 + r) * 32], 0, g), o0);
    o1 = wmma16(pf, load_fragB(&vbuf[cur][(1 * 16 + r) * 32], 0, g), o1);
    o2 = wmma16(pf, load_fragB(&vbuf[cur][(2 * 16 + r) * 32], 0, g), o2);
    o3 = wmma16(pf, load_fragB(&vbuf[cur][(3 * 16 + r) * 32], 0, g), o3);
    __syncthreads();      // all waves done reading buf[cur] before its restage
  }

#pragma unroll
  for (int j = 0; j < 8; ++j) {
    const float inv = 1.0f / lrow[j];
    const size_t row = (size_t)(b * N_ + q0 + j + g * 8) * C_ + h * D_;
    aoh[row + 0 * 16 + r] = (half_t)(o0[j] * inv);
    aoh[row + 1 * 16 + r] = (half_t)(o1[j] * inv);
    aoh[row + 2 * 16 + r] = (half_t)(o2[j] * inv);
    aoh[row + 3 * 16 + r] = (half_t)(o3[j] * inv);
  }
}

// ------------------------------------------------- output projection GEMM
__global__ __launch_bounds__(128) void proj_gemm(const half_t* __restrict__ ah,
                                                 const half_t* __restrict__ wh,
                                                 const float* __restrict__ bias,
                                                 float* __restrict__ out) {
  const int lane = threadIdx.x & 31;
  const int wave = threadIdx.x >> 5;
  const int g = lane >> 4;
  const int r = lane & 15;
  const int NT = C_ / 32;                       // 32 column tiles
  const int tile = blockIdx.x * 4 + wave;
  const int tm = (tile / NT) * 32;
  const int tn = (tile % NT) * 32;

  v8f acc[2][2] = {};
  const half_t* a0p = ah + (size_t)(tm + r) * C_;
  const half_t* a1p = ah + (size_t)(tm + 16 + r) * C_;
  const half_t* b0p = wh + (size_t)(tn + r) * C_;
  const half_t* b1p = wh + (size_t)(tn + 16 + r) * C_;
  for (int k = 0; k < C_; k += 32) {
    const v16h a0 = load_fragA(a0p, k, g);
    const v16h a1 = load_fragA(a1p, k, g);
    const v16h b0 = load_fragB(b0p, k, g);
    const v16h b1 = load_fragB(b1p, k, g);
    acc[0][0] = wmma16(a0, b0, acc[0][0]);
    acc[0][1] = wmma16(a0, b1, acc[0][1]);
    acc[1][0] = wmma16(a1, b0, acc[1][0]);
    acc[1][1] = wmma16(a1, b1, acc[1][1]);
  }
#pragma unroll
  for (int mi = 0; mi < 2; ++mi)
#pragma unroll
    for (int ni = 0; ni < 2; ++ni) {
      const int col = tn + ni * 16 + r;
      const float bv = bias[col];
      const int m0 = tm + mi * 16 + g * 8;
#pragma unroll
      for (int j = 0; j < 8; ++j)
        out[(size_t)(m0 + j) * C_ + col] = acc[mi][ni][j] + bv;
    }
}

// ---------------------------------------------------------------- launcher
extern "C" void kernel_launch(void* const* d_in, const int* in_sizes, int n_in,
                              void* d_out, int out_size, void* d_ws, size_t ws_size,
                              hipStream_t stream) {
  const float* x         = (const float*)d_in[0];
  const float* attn_bias = (const float*)d_in[1];
  const float* qkv_w     = (const float*)d_in[2];
  const float* qkv_b     = (const float*)d_in[3];
  const float* proj_w    = (const float*)d_in[4];
  const float* proj_b    = (const float*)d_in[5];
  float* out = (float*)d_out;

  char* ws = (char*)d_ws;                       // 88 MB used
  half_t* xh     = (half_t*)(ws);                         // 16 MB
  half_t* wqkvh  = (half_t*)(ws + (size_t)16 * 1048576);  //  6 MB
  half_t* wprojh = (half_t*)(ws + (size_t)22 * 1048576);  //  2 MB
  half_t* qh     = (half_t*)(ws + (size_t)24 * 1048576);  // 16 MB [B,H,N,D]
  half_t* kh     = (half_t*)(ws + (size_t)40 * 1048576);  // 16 MB [B,H,N,D]
  half_t* vth    = (half_t*)(ws + (size_t)56 * 1048576);  // 16 MB [B,H,D,N]
  half_t* aoh    = (half_t*)(ws + (size_t)72 * 1048576);  // 16 MB [B,N,C]

  cvt_f32_to_f16<<<8192, 256, 0, stream>>>(x,      xh,     (B_ * N_ * C_) / 4);
  cvt_f32_to_f16<<<3072, 256, 0, stream>>>(qkv_w,  wqkvh,  (3 * C_ * C_) / 4);
  cvt_f32_to_f16<<<1024, 256, 0, stream>>>(proj_w, wprojh, (C_ * C_) / 4);

  // 8192x3072 output tiles of 32x32: 24576 wave-tiles / 4 waves per block
  qkv_gemm<<<6144, 128, 0, stream>>>(xh, wqkvh, qkv_b, qh, kh, vth);
  // 64 (b,h) pairs x 16 blocks of 128 query rows
  flash_attn<<<1024, 256, 0, stream>>>(qh, kh, vth, attn_bias, aoh);
  // 8192x1024 output tiles of 32x32: 8192 wave-tiles / 4 per block
  proj_gemm<<<2048, 128, 0, stream>>>(aoh, wprojh, proj_b, out);
}
